// S4ModNModel_8418135900413
// MI455X (gfx1250) — compile-verified, matrix operand order checked by reading, not run on hardware
//
#include <hip/hip_runtime.h>
#include <hip/hip_bf16.h>
#include <math.h>

// ---- problem constants (match reference) ----
constexpr int Hd = 1024, Nd = 64, Ld = 2, Bd = 32, Td = 512, STEPSd = 16;
constexpr int H2 = 2 * Hd;          // 2048
constexpr int BT = Bd * Td;         // 16384
constexpr int HN = Hd * Nd;         // 65536

// ---- CDNA5 async global->LDS path (probe via __has_builtin) ----
#if defined(__has_builtin)
#  if __has_builtin(__builtin_amdgcn_global_load_async_to_lds_b128) && \
      __has_builtin(__builtin_amdgcn_s_wait_asynccnt)
#    define USE_ASYNC_LDS 1
#  endif
#endif
#ifndef USE_ASYNC_LDS
#  define USE_ASYNC_LDS 0
#endif

#if USE_ASYNC_LDS
// builtin signature (from hipcc diagnostics / LLVM "vV4i*1V4i*3IiIi"):
//   void __builtin_amdgcn_global_load_async_to_lds_b128(
//       v4i addrspace(1)* gsrc, v4i addrspace(3)* ldst, imm int off, imm int cpol)
typedef int async_v4i __attribute__((vector_size(4 * sizeof(int))));
typedef __attribute__((address_space(1))) async_v4i* gptr_v4i;
typedef __attribute__((address_space(3))) async_v4i* lptr_v4i;
#endif

// ---- WMMA vector types (CDNA5 gfx1250, wave32) ----
typedef __attribute__((ext_vector_type(16))) _Float16 v16h;
typedef __attribute__((ext_vector_type(8)))  float    v8f;

union V8F  { v8f v;  float f[8]; };
union V16H { v16h v; unsigned int u[8]; _Float16 h[16]; };

__device__ __forceinline__ float gelu_exact(float x) {
    return 0.5f * x * (1.0f + erff(x * 0.70710678118654752f));
}
__device__ __forceinline__ float sigmoidf(float x) {
    return 1.0f / (1.0f + expf(-x));
}

// ------------------------------------------------------------------
// dA = exp(dt*A), dB = (dA-1)/A * B   (complex, diagonal) per (l,h,n)
// ------------------------------------------------------------------
__global__ void precompute_kernel(const float* __restrict__ log_dt,
                                  const float* __restrict__ Ar, const float* __restrict__ Ai,
                                  const float* __restrict__ Br, const float* __restrict__ Bi,
                                  float* __restrict__ dAr, float* __restrict__ dAi,
                                  float* __restrict__ dBr, float* __restrict__ dBi) {
    int idx = blockIdx.x * 256 + threadIdx.x;
    if (idx >= Ld * HN) return;
    int lh = idx / Nd;
    float dt = expf(log_dt[lh]);
    float ar = Ar[idx], ai = Ai[idx];
    float e  = expf(dt * ar);
    float car = e * cosf(dt * ai);
    float cai = e * sinf(dt * ai);
    dAr[idx] = car; dAi[idx] = cai;
    float nr = car - 1.0f, ni = cai;
    float inv = 1.0f / (ar * ar + ai * ai);
    float qr = (nr * ar + ni * ai) * inv;
    float qi = (ni * ar - nr * ai) * inv;
    float br = Br[idx], bi = Bi[idx];
    dBr[idx] = qr * br - qi * bi;
    dBi[idx] = qr * bi + qi * br;
}

__global__ void convw_kernel(const float* __restrict__ w, _Float16* __restrict__ wh) {
    int idx = blockIdx.x * 256 + threadIdx.x;   // L*H*2H elements, exact grid
    wh[idx] = (_Float16)w[idx];
}

__global__ void in_proj_kernel(const float* __restrict__ u, const float* __restrict__ w,
                               const float* __restrict__ b, float* __restrict__ x) {
    int idx = blockIdx.x * 256 + threadIdx.x;   // B*T*H, exact grid
    int h  = idx & (Hd - 1);
    int bt = idx >> 10;
    x[idx] = u[bt] * w[h] + b[h];
}

__global__ void fill_kernel(float* __restrict__ p, int n) {
    int idx = blockIdx.x * 256 + threadIdx.x;
    if (idx < n) p[idx] = 0.0f;
}

__global__ void context_kernel(const float* __restrict__ x, float* __restrict__ ctx) {
    int idx = blockIdx.x * 256 + threadIdx.x;   // B*H
    int b = idx >> 10, h = idx & (Hd - 1);
    ctx[idx] = x[((size_t)(b * Td + Td - 1)) * Hd + h];
}

__global__ void broadcast_kernel(const float* __restrict__ din,
                                 float* __restrict__ u0, float* __restrict__ dec) {
    int idx = blockIdx.x * 256 + threadIdx.x;   // B*H
    float v = din[idx >> 10];
    u0[idx] = v; dec[idx] = v;
}

// ------------------------------------------------------------------
// SSM scan: one wave32 per (b,h); 2 complex states per lane (N=64),
// cross-lane reduce per timestep via shfl_xor. Latency-bound, so
// prefetch the strided u-stream a few iterations ahead.
// ------------------------------------------------------------------
__global__ __launch_bounds__(256) void scan_kernel(
    const float* __restrict__ x,
    const float* __restrict__ dAr, const float* __restrict__ dAi,
    const float* __restrict__ dBr, const float* __restrict__ dBi,
    const float* __restrict__ Cr,  const float* __restrict__ Ci,
    const float* __restrict__ Dm,  float* __restrict__ y) {
    int w    = blockIdx.x * 8 + (threadIdx.x >> 5);
    int lane = threadIdx.x & 31;
    int b = w >> 10, h = w & (Hd - 1);
    int p0 = h * Nd + lane * 2;
    float a0r = dAr[p0], a0i = dAi[p0], a1r = dAr[p0 + 1], a1i = dAi[p0 + 1];
    float b0r = dBr[p0], b0i = dBi[p0], b1r = dBr[p0 + 1], b1i = dBi[p0 + 1];
    float c0r = Cr[p0],  c0i = Ci[p0],  c1r = Cr[p0 + 1],  c1i = Ci[p0 + 1];
    float Dh = Dm[h];
    float s0r = 0.f, s0i = 0.f, s1r = 0.f, s1i = 0.f;
    const float* xp = x + (size_t)b * Td * Hd + h;
    float*       yp = y + (size_t)b * Td * Hd + h;
    for (int t = 0; t < Td; ++t) {
        if (t + 8 < Td)
            __builtin_prefetch((const void*)(xp + (size_t)(t + 8) * Hd), 0, 1);
        float u = xp[(size_t)t * Hd];
        float t0r = a0r * s0r - a0i * s0i + b0r * u;
        float t0i = a0r * s0i + a0i * s0r + b0i * u;
        float t1r = a1r * s1r - a1i * s1i + b1r * u;
        float t1i = a1r * s1i + a1i * s1r + b1i * u;
        s0r = t0r; s0i = t0i; s1r = t1r; s1i = t1i;
        float p = c0r * s0r - c0i * s0i + c1r * s1r - c1i * s1i;
        p += __shfl_xor(p, 16, 32);
        p += __shfl_xor(p, 8, 32);
        p += __shfl_xor(p, 4, 32);
        p += __shfl_xor(p, 2, 32);
        p += __shfl_xor(p, 1, 32);
        if (lane == 0) yp[(size_t)t * Hd] = 2.0f * p + Dh * u;
    }
}

// single-step recurrence for decode; states persist in global scratch
__global__ __launch_bounds__(256) void dec_recur_kernel(
    const float* __restrict__ dec,
    float* __restrict__ sre, float* __restrict__ sim,
    const float* __restrict__ dAr, const float* __restrict__ dAi,
    const float* __restrict__ dBr, const float* __restrict__ dBi,
    const float* __restrict__ Cr,  const float* __restrict__ Ci,
    const float* __restrict__ Dm,  float* __restrict__ ydec) {
    int w    = blockIdx.x * 8 + (threadIdx.x >> 5);
    int lane = threadIdx.x & 31;
    int b = w >> 10, h = w & (Hd - 1);
    int p0 = h * Nd + lane * 2;
    int ps = ((b << 10) + h) * Nd + lane * 2;
    float u = dec[(b << 10) + h];
    float a0r = dAr[p0], a0i = dAi[p0], a1r = dAr[p0 + 1], a1i = dAi[p0 + 1];
    float b0r = dBr[p0], b0i = dBi[p0], b1r = dBr[p0 + 1], b1i = dBi[p0 + 1];
    float c0r = Cr[p0],  c0i = Ci[p0],  c1r = Cr[p0 + 1],  c1i = Ci[p0 + 1];
    float s0r = sre[ps], s0i = sim[ps], s1r = sre[ps + 1], s1i = sim[ps + 1];
    float t0r = a0r * s0r - a0i * s0i + b0r * u;
    float t0i = a0r * s0i + a0i * s0r + b0i * u;
    float t1r = a1r * s1r - a1i * s1i + b1r * u;
    float t1i = a1r * s1i + a1i * s1r + b1i * u;
    sre[ps] = t0r; sim[ps] = t0i; sre[ps + 1] = t1r; sim[ps + 1] = t1i;
    float p = c0r * t0r - c0i * t0i + c1r * t1r - c1i * t1i;
    p += __shfl_xor(p, 16, 32);
    p += __shfl_xor(p, 8, 32);
    p += __shfl_xor(p, 4, 32);
    p += __shfl_xor(p, 2, 32);
    p += __shfl_xor(p, 1, 32);
    if (lane == 0) ydec[(b << 10) + h] = 2.0f * p + Dm[h] * u;
}

// ------------------------------------------------------------------
// Fused GEMM: out = GLU(gelu(Y) @ W + bias).
// Y: M x 1024 (f32, raw), W: 1024 x 2048 (f16), out: M x 1024 (f32).
// Block = 256 thr (8 waves, 2x4). Block tile: 32 rows x 64 cols, with the
// paired "+H" columns computed simultaneously so GLU fuses into epilogue.
// Each wave accumulates two 16x16 f32 tiles via v_wmma_f32_16x16x32_f16.
// B tiles are staged global->LDS with ASYNCcnt-tracked async copies
// (global_load_async_to_lds_b128 + s_wait_asynccnt) when available.
// ------------------------------------------------------------------
__global__ __launch_bounds__(256) void gemm_gelu_glu_kernel(
    const float* __restrict__ Y, const _Float16* __restrict__ W,
    const float* __restrict__ bias, float* __restrict__ out, int M) {
    __shared__ __align__(16) _Float16 As[32 * 32];   // [row][k]
    __shared__ __align__(16) _Float16 Blo[32 * 64];  // [k][n]  (async-copy friendly)
    __shared__ __align__(16) _Float16 Bhi[32 * 64];  // [k][n]

    const int tid  = threadIdx.x;
    const int lane = tid & 31;
    const int wv   = tid >> 5;       // 0..7
    const int wm   = wv >> 2;        // 0..1
    const int wn   = wv & 3;         // 0..3
    const int m0   = blockIdx.x * 32;
    const int n0   = blockIdx.y * 64;

    V8F clo, chi;
#pragma unroll
    for (int r = 0; r < 8; ++r) { clo.f[r] = 0.0f; chi.f[r] = 0.0f; }

    const int mrow = wm * 16 + (lane & 15);
    const int ncol = wn * 16 + (lane & 15);
    const int kb   = (lane >> 4) * 8;

    // per-thread B staging coordinates: one 16B chunk per tile per thread
    const int bk = tid >> 3;          // 0..31 (k within tile)
    const int bn = (tid & 7) * 8;     // 0,8,...,56 (n within tile)

    for (int k0 = 0; k0 < Hd; k0 += 32) {
        // kick off async B-tile copies first so they overlap the GELU staging
        {
            const _Float16* gLo = W + (size_t)(k0 + bk) * H2 + (n0 + bn);
            const _Float16* gHi = gLo + Hd;
#if USE_ASYNC_LDS
            __builtin_amdgcn_global_load_async_to_lds_b128(
                (gptr_v4i)gLo, (lptr_v4i)&Blo[bk * 64 + bn], 0, 0);
            __builtin_amdgcn_global_load_async_to_lds_b128(
                (gptr_v4i)gHi, (lptr_v4i)&Bhi[bk * 64 + bn], 0, 0);
#else
#pragma unroll
            for (int q = 0; q < 8; ++q) {
                Blo[bk * 64 + bn + q] = gLo[q];
                Bhi[bk * 64 + bn + q] = gHi[q];
            }
#endif
        }
        // stage A tile (32x32) with fused exact-GELU + f32->f16
#pragma unroll
        for (int q = 0; q < 4; ++q) {
            int idx = q * 256 + tid;        // 0..1023
            int r = idx >> 5, k = idx & 31;
            float v = 0.0f;
            int row = m0 + r;
            if (row < M) v = Y[(size_t)row * Hd + (k0 + k)];
            As[r * 32 + k] = (_Float16)gelu_exact(v);
        }
#if USE_ASYNC_LDS
        __builtin_amdgcn_s_wait_asynccnt(0);
#endif
        __syncthreads();

        // build fragments per ISA 16-bit A 16x32 layout (K-pairs per dword)
        V16H a, bl, bh;
        const unsigned int* As_u = (const unsigned int*)As;
#pragma unroll
        for (int vv = 0; vv < 8; ++vv) {
            int kk = (vv < 4 ? 2 * vv : 16 + 2 * (vv - 4)) + kb;  // even
            a.u[vv] = As_u[(mrow * 32 + kk) >> 1];
            // B stored [k][n]: gather the K-pair as two 16-bit LDS reads
            bl.h[2 * vv]     = Blo[kk * 64 + ncol];
            bl.h[2 * vv + 1] = Blo[(kk + 1) * 64 + ncol];
            bh.h[2 * vv]     = Bhi[kk * 64 + ncol];
            bh.h[2 * vv + 1] = Bhi[(kk + 1) * 64 + ncol];
        }
        clo.v = __builtin_amdgcn_wmma_f32_16x16x32_f16(false, a.v, false, bl.v,
                                                       (short)0, clo.v, false, false);
        chi.v = __builtin_amdgcn_wmma_f32_16x16x32_f16(false, a.v, false, bh.v,
                                                       (short)0, chi.v, false, false);
        __syncthreads();
    }

    // epilogue: bias + GLU  (C/D layout: VGPR r -> M=r(+8), N=lane&15)
    const int colin = lane & 15;
    const int rbase = (lane >> 4) * 8;
    const int col   = n0 + wn * 16 + colin;
    const float ba  = bias[col];
    const float bb  = bias[Hd + col];
#pragma unroll
    for (int r = 0; r < 8; ++r) {
        int row = m0 + wm * 16 + r + rbase;
        if (row < M) {
            float av = clo.f[r] + ba;
            float bv = chi.f[r] + bb;
            out[(size_t)row * Hd + col] = av * sigmoidf(bv);
        }
    }
}

// LayerNorm over H of (gin + res), affine, one block per row
__global__ __launch_bounds__(256) void ln_kernel(
    const float* __restrict__ gin, const float* __restrict__ res,
    const float* __restrict__ gam, const float* __restrict__ bet,
    float* __restrict__ outp) {
    __shared__ float sb[256], sb2[256];
    int row = blockIdx.x, tid = threadIdx.x;
    const float* gp = gin + (size_t)row * Hd;
    const float* rp = res + (size_t)row * Hd;
    float v[4]; float s = 0.f, s2 = 0.f;
#pragma unroll
    for (int q = 0; q < 4; ++q) {
        int c = q * 256 + tid;
        float t = gp[c] + rp[c];
        v[q] = t; s += t; s2 += t * t;
    }
    sb[tid] = s; sb2[tid] = s2; __syncthreads();
    for (int st = 128; st > 0; st >>= 1) {
        if (tid < st) { sb[tid] += sb[tid + st]; sb2[tid] += sb2[tid + st]; }
        __syncthreads();
    }
    float m   = sb[0] * (1.0f / Hd);
    float var = sb2[0] * (1.0f / Hd) - m * m;
    float inv = rsqrtf(var + 1e-5f);
    float* op = outp + (size_t)row * Hd;
#pragma unroll
    for (int q = 0; q < 4; ++q) {
        int c = q * 256 + tid;
        op[c] = (v[q] - m) * inv * gam[c] + bet[c];
    }
}

// head: bit = sigmoid((dec+ctx) @ head_w + head_b); one block per batch row
__global__ __launch_bounds__(256) void head_kernel(
    const float* __restrict__ dec, const float* __restrict__ ctx,
    const float* __restrict__ hw, const float* __restrict__ hb,
    float* __restrict__ out, float* __restrict__ din, int stepi) {
    __shared__ float sb[256];
    int b = blockIdx.x, tid = threadIdx.x;
    float s = 0.f;
#pragma unroll
    for (int q = 0; q < 4; ++q) {
        int h = q * 256 + tid;
        s += (dec[b * Hd + h] + ctx[b * Hd + h]) * hw[h];
    }
    sb[tid] = s; __syncthreads();
    for (int st = 128; st > 0; st >>= 1) {
        if (tid < st) sb[tid] += sb[tid + st];
        __syncthreads();
    }
    if (tid == 0) {
        float bit = sigmoidf(sb[0] + hb[0]);
        out[b * STEPSd + stepi] = bit;
        din[b] = bit;
    }
}

// ------------------------------------------------------------------
extern "C" void kernel_launch(void* const* d_in, const int* in_sizes, int n_in,
                              void* d_out, int out_size, void* d_ws, size_t ws_size,
                              hipStream_t stream) {
    (void)in_sizes; (void)n_in; (void)out_size; (void)ws_size;
    const float* input_seq = (const float*)d_in[0];
    // d_in[1] = autoregressive_steps (int scalar); compile-time STEPSd used
    const float* in_w   = (const float*)d_in[2];
    const float* in_b   = (const float*)d_in[3];
    const float* log_dt = (const float*)d_in[4];
    const float* A_re   = (const float*)d_in[5];
    const float* A_im   = (const float*)d_in[6];
    const float* B_re   = (const float*)d_in[7];
    const float* B_im   = (const float*)d_in[8];
    const float* C_re   = (const float*)d_in[9];
    const float* C_im   = (const float*)d_in[10];
    const float* Dm     = (const float*)d_in[11];
    const float* out_w  = (const float*)d_in[12];
    const float* out_b  = (const float*)d_in[13];
    const float* ln_g   = (const float*)d_in[14];
    const float* ln_b   = (const float*)d_in[15];
    const float* head_w = (const float*)d_in[16];
    const float* head_b = (const float*)d_in[17];
    float* out = (float*)d_out;

    // workspace carve (256B aligned)
    char* p = (char*)d_ws;
    auto carve = [&](size_t bytes) -> void* {
        void* r = (void*)p;
        p += (bytes + 255) & ~(size_t)255;
        return r;
    };
    float*    dAr   = (float*)carve(sizeof(float) * Ld * HN);
    float*    dAi   = (float*)carve(sizeof(float) * Ld * HN);
    float*    dBr   = (float*)carve(sizeof(float) * Ld * HN);
    float*    dBi   = (float*)carve(sizeof(float) * Ld * HN);
    float*    x     = (float*)carve(sizeof(float) * (size_t)BT * Hd);
    float*    ybuf  = (float*)carve(sizeof(float) * (size_t)BT * Hd);
    float*    gbuf  = (float*)carve(sizeof(float) * (size_t)BT * Hd);
    _Float16* whalf = (_Float16*)carve(sizeof(_Float16) * (size_t)Ld * Hd * H2);
    float*    ctx   = (float*)carve(sizeof(float) * Bd * Hd);
    float*    sre   = (float*)carve(sizeof(float) * (size_t)Ld * Bd * HN);
    float*    sim   = (float*)carve(sizeof(float) * (size_t)Ld * Bd * HN);
    float*    u0    = (float*)carve(sizeof(float) * Bd * Hd);
    float*    dec   = (float*)carve(sizeof(float) * Bd * Hd);
    float*    ydec  = (float*)carve(sizeof(float) * Bd * Hd);
    float*    gdec  = (float*)carve(sizeof(float) * Bd * Hd);
    float*    din   = (float*)carve(sizeof(float) * Bd);

    // precompute discretized SSM params + f16 weights
    precompute_kernel<<<(Ld * HN + 255) / 256, 256, 0, stream>>>(
        log_dt, A_re, A_im, B_re, B_im, dAr, dAi, dBr, dBi);
    convw_kernel<<<(Ld * Hd * H2) / 256, 256, 0, stream>>>(out_w, whalf);
    in_proj_kernel<<<((size_t)BT * Hd) / 256, 256, 0, stream>>>(input_seq, in_w, in_b, x);

    // prefill: 2 residual S4 blocks
    for (int j = 0; j < Ld; ++j) {
        scan_kernel<<<(Bd * Hd) / 8, 256, 0, stream>>>(
            x, dAr + j * HN, dAi + j * HN, dBr + j * HN, dBi + j * HN,
            C_re + j * HN, C_im + j * HN, Dm + j * Hd, ybuf);
        gemm_gelu_glu_kernel<<<dim3(BT / 32, Hd / 64), 256, 0, stream>>>(
            ybuf, whalf + (size_t)j * Hd * H2, out_b + j * H2, gbuf, BT);
        ln_kernel<<<BT, 256, 0, stream>>>(gbuf, x, ln_g + j * Hd, ln_b + j * Hd, x);
    }
    context_kernel<<<(Bd * Hd) / 256, 256, 0, stream>>>(x, ctx);

    // zero decode states + initial carry
    fill_kernel<<<((size_t)Ld * Bd * HN + 255) / 256, 256, 0, stream>>>(sre, Ld * Bd * HN);
    fill_kernel<<<((size_t)Ld * Bd * HN + 255) / 256, 256, 0, stream>>>(sim, Ld * Bd * HN);
    fill_kernel<<<1, 256, 0, stream>>>(din, Bd);

    // autoregressive decode
    for (int i = 0; i < STEPSd; ++i) {
        broadcast_kernel<<<(Bd * Hd) / 256, 256, 0, stream>>>(din, u0, dec);
        for (int j = 0; j < Ld; ++j) {
            dec_recur_kernel<<<(Bd * Hd) / 8, 256, 0, stream>>>(
                dec, sre + (size_t)j * Bd * HN, sim + (size_t)j * Bd * HN,
                dAr + j * HN, dAi + j * HN, dBr + j * HN, dBi + j * HN,
                C_re + j * HN, C_im + j * HN, Dm + j * Hd, ydec);
            gemm_gelu_glu_kernel<<<dim3(1, Hd / 64), 256, 0, stream>>>(
                ydec, whalf + (size_t)j * Hd * H2, out_b + j * H2, gdec, Bd);
            ln_kernel<<<Bd, 256, 0, stream>>>(
                gdec, (i == 0) ? u0 : gdec, ln_g + j * Hd, ln_b + j * Hd, dec);
        }
        head_kernel<<<Bd, 256, 0, stream>>>(dec, ctx, head_w, head_b, out, din, i);
    }
}
